// ReCon_89215060673074
// MI455X (gfx1250) — compile-verified, matrix operand order checked by reading, not run on hardware
//
#include <hip/hip_runtime.h>
#include <hip/hip_bf16.h>
#include <math.h>
#include <stdint.h>

typedef __bf16 bf16;
typedef unsigned int u32;
typedef __attribute__((ext_vector_type(16))) __bf16 v16bf;
typedef __attribute__((ext_vector_type(8)))  float  v8f;
typedef int i32x4 __attribute__((ext_vector_type(4)));
typedef int i32x8 __attribute__((ext_vector_type(8)));

#define BATCH 64
#define NP 256
#define DIM 512
#define DEPTH 6
#define NHEAD 8
#define DHEAD 64
#define MLPD 2048
#define MROWS (BATCH * NP)   // 16384
#define EPSV 1e-5f

// ---------------------------------------------------------------------------
// Fragment builder: 16-bit WMMA A/B fragment from a row whose K elements are
// contiguous. ISA 16x32 layout:
//   lanes 0-15 : elems j=0..7 -> K=j,    j=8..15 -> K=16+(j-8)
//   lanes 16-31: elems j=0..7 -> K=8+j,  j=8..15 -> K=24+(j-8)
// dword-pair i=0..3 -> pair 4*ksel+i, i=4..7 -> 8+4*ksel+(i-4).
// ---------------------------------------------------------------------------
__device__ inline v16bf frag16(const u32* rowp, int ksel) {
  union { v16bf v; u32 u[8]; } f;
#pragma unroll
  for (int i = 0; i < 4; ++i) {
    f.u[i]     = rowp[4 * ksel + i];
    f.u[4 + i] = rowp[8 + 4 * ksel + i];
  }
  return f.v;
}

// ---------------------------------------------------------------------------
// Tensor Data Mover: DMA a 2D bf16 tile (tile_w x tile_h elements) from global
// (row stride stride0 elements) into LDS at lds_off. D# per ISA 08 sec 8.3/8.4:
//  group0: [1:0]=count=1, [63:32]=lds_addr, [120:64]=global_addr, [127:126]=2
//  group1: data_size=1(2B); tensor_dim0/1; tile_dim0/1; tensor_dim0_stride
// Issued once per wave; completion tracked by TENSORcnt.
// ---------------------------------------------------------------------------
__device__ inline void tdm_load_2d_bf16(u32 lds_off, const void* gptr,
                                        u32 tile_w, u32 tile_h,
                                        u32 tdim0, u32 tdim1, u32 stride0) {
  unsigned long long ga = (unsigned long long)(uintptr_t)gptr;
  i32x4 g0;
  g0.x = 1;                                   // user descriptor, count=1
  g0.y = (int)lds_off;                        // D#.lds_addr
  g0.z = (int)(u32)ga;                        // global_addr[31:0]
  g0.w = (int)(((u32)(ga >> 32) & 0x01FFFFFFu) | 0x80000000u);  // [56:32], type=2
  i32x8 g1;
  g1.s0 = (int)0x00010000u;                   // workgroup_mask=0, data_size=2B
  g1.s1 = (int)((tdim0 & 0xFFFFu) << 16);                         // dim0 lo
  g1.s2 = (int)(((tdim0 >> 16) & 0xFFFFu) | ((tdim1 & 0xFFFFu) << 16));
  g1.s3 = (int)(((tdim1 >> 16) & 0xFFFFu) | (tile_w << 16));      // tile_dim0
  g1.s4 = (int)(tile_h & 0xFFFFu);                                // tile_dim1
  g1.s5 = (int)stride0;                                           // dim0_stride
  g1.s6 = 0;
  g1.s7 = 0;
  asm volatile("tensor_load_to_lds %0, %1" :: "s"(g0), "s"(g1) : "memory");
}

// ---------------------------------------------------------------------------
// Generic bf16 GEMM  C(MxN) = A(MxK) * B(KxN), WMMA f32 accumulate.
// Block tile 128x128, 8 waves (4 in M, 2 in N), wave tile 32x64 (2x4 frags).
// A tile (128x32) is fetched by the Tensor Data Mover (wave 0 issues, all
// waves consume after s_wait_tensorcnt + barrier). B tile staged transposed.
// EPI: 0 = store bf16 C
//      1 = Cf[r,c] += acc + bias[c]        (residual add into fp32 h)
//      2 = store bf16 gelu(acc + bias[c])  (exact erf GELU)
//      3 = ct1 epilogue: leaky(acc+bias[o]) scattered to ct2-input layout
// ---------------------------------------------------------------------------
template <int EPI>
__global__ void __launch_bounds__(256) gemm_bf16(
    const bf16* __restrict__ A, const bf16* __restrict__ Bw,
    int M, int N, int K,
    float* __restrict__ Cf, bf16* __restrict__ Cb,
    const float* __restrict__ bias) {
  __shared__ __align__(16) bf16 As[128 * 32];  // TDM-filled, row = 32 halves (64B)
  __shared__ __align__(16) bf16 Bt[128][40];   // B transposed: [n][k], 80B stride

  const int tid  = threadIdx.x;
  const int lane = tid & 31;
  const int wid  = tid >> 5;
  const int wM   = wid & 3;       // 0..3
  const int wN   = wid >> 2;      // 0..1
  const int bN   = blockIdx.x * 128;
  const int bM   = blockIdx.y * 128;
  const int ksel = lane >> 4;
  const int r    = lane & 15;

  v8f acc[2][4];
#pragma unroll
  for (int tm = 0; tm < 2; ++tm)
#pragma unroll
    for (int tn = 0; tn < 4; ++tn)
#pragma unroll
      for (int i = 0; i < 8; ++i) acc[tm][tn][i] = 0.0f;

  for (int kk = 0; kk < K; kk += 32) {
    // ---- A tile via Tensor Data Mover (one instruction, wave 0)
    if (wid == 0)
      tdm_load_2d_bf16((u32)(uintptr_t)As, A + (size_t)bM * K + kk,
                       /*tile_w=*/32, /*tile_h=*/128,
                       /*tdim0=*/(u32)(K - kk), /*tdim1=*/(u32)(M - bM),
                       /*stride0=*/(u32)K);
    // ---- stage B tile transposed: Bt[n][k] (all 256 threads)
#pragma unroll
    for (int it = 0; it < 2; ++it) {
      int c   = tid + it * 256;
      int k   = c & 31;
      int nch = c >> 5;                // 0..15, 8 columns each
      union { uint4 u4; bf16 h[8]; } tmp;
      tmp.u4 = *(const uint4*)(Bw + (size_t)(kk + k) * N + bN + nch * 8);
#pragma unroll
      for (int i = 0; i < 8; ++i) Bt[nch * 8 + i][k] = tmp.h[i];
    }
    if (kk + 32 < K)  // warm next B tile rows (global_prefetch_b8)
      __builtin_prefetch((const void*)(Bw + (size_t)(kk + 32 + (tid >> 3)) * N + bN), 0, 0);
    if (wid == 0) __builtin_amdgcn_s_wait_tensorcnt(0);
    __syncthreads();

    v16bf af[2], bfr[4];
#pragma unroll
    for (int tm = 0; tm < 2; ++tm)
      af[tm] = frag16((const u32*)(As + (wM * 32 + tm * 16 + r) * 32), ksel);
#pragma unroll
    for (int tn = 0; tn < 4; ++tn)
      bfr[tn] = frag16((const u32*)&Bt[wN * 64 + tn * 16 + r][0], ksel);
#pragma unroll
    for (int tm = 0; tm < 2; ++tm)
#pragma unroll
      for (int tn = 0; tn < 4; ++tn)
        acc[tm][tn] = __builtin_amdgcn_wmma_f32_16x16x32_bf16(
            false, af[tm], false, bfr[tn], (short)0, acc[tm][tn], false, false);
    __syncthreads();
  }

  // ---- epilogue.  C layout: elem i of acc -> (m = i + 8*(lane>>4), n = lane&15)
  const int hlf = lane >> 4;
  const int nl  = lane & 15;
#pragma unroll
  for (int tm = 0; tm < 2; ++tm)
#pragma unroll
    for (int tn = 0; tn < 4; ++tn)
#pragma unroll
      for (int i = 0; i < 8; ++i) {
        int   row = bM + wM * 32 + tm * 16 + i + 8 * hlf;
        int   col = bN + wN * 64 + tn * 16 + nl;
        float v   = acc[tm][tn][i];
        if (EPI == 0) {
          Cb[(size_t)row * N + col] = (bf16)v;
        } else if (EPI == 1) {
          size_t o = (size_t)row * N + col;
          Cf[o] = Cf[o] + v + bias[col];
        } else if (EPI == 2) {
          float g = v + bias[col];
          g = 0.5f * g * (1.0f + erff(g * 0.70710678118654752f));
          Cb[(size_t)row * N + col] = (bf16)g;
        } else {  // EPI == 3 : ConvTranspose(4,4) scatter, LeakyReLU(0.2)
          int   b_ = row >> 8, s_ = row & 255;        // s = i*16 + j on 16x16 grid
          int   oc = col >> 4, pp = (col >> 2) & 3, qq = col & 3;
          float g  = v + bias[oc];
          g = (g >= 0.0f) ? g : 0.2f * g;
          int y  = 4 * (s_ >> 4) + pp;
          int x2 = 4 * (s_ & 15) + qq;
          Cb[((size_t)b_ * 4096 + y * 64 + x2) * 64 + oc] = (bf16)g;
        }
      }
}

// ---------------------------------------------------------------------------
// LSA attention, one (batch, head, half-of-rows) per workgroup, 8 waves.
// Each wave: 16 query rows; 16 score tiles kept in registers; diag mask;
// learned temperature; row softmax via shfl_xor in the 16-lane half; P@V via
// per-wave LDS staging of P fragments. K then V^T share one 32KB LDS region.
// ---------------------------------------------------------------------------
__global__ void __launch_bounds__(256) attn_kernel(
    const bf16* __restrict__ qkv, const float* __restrict__ temp, int layer,
    bf16* __restrict__ outb) {
  __shared__ __align__(16) bf16 KV[256 * 64];      // Ks[m][d] then Vt[d][m]
  __shared__ __align__(16) bf16 ps[8][16][32];     // per-wave P staging

  const int tid   = threadIdx.x;
  const int lane  = tid & 31;
  const int wid   = tid >> 5;
  const int wg    = blockIdx.x;
  const int halfb = wg & 1;
  const int bh    = wg >> 1;
  const int b     = bh >> 3;
  const int hh    = bh & 7;
  const int qrow0 = halfb * 128 + wid * 16;
  const int ksel  = lane >> 4;
  const int r     = lane & 15;
  const int hlf   = lane >> 4;
  const int nl    = lane & 15;
  const float tscale = __expf(temp[layer]);

  // ---- stage K rows: Ks[m][d], 64B (d = 0..31) of a row per thread
  {
    const uint4* src = (const uint4*)(qkv + (size_t)(b * NP + tid) * 1536 + 512 + hh * 64);
    uint4*       dst = (uint4*)&KV[tid * 64];
#pragma unroll
    for (int i = 0; i < 4; ++i) dst[i] = src[i];
  }
  __syncthreads();

  // ---- q fragments straight from global (bf16, contiguous in d)
  const u32* qp = (const u32*)(qkv + (size_t)(b * NP + qrow0 + r) * 1536 + hh * 64);
  v16bf aq0 = frag16(qp, ksel);
  v16bf aq1 = frag16(qp + 16, ksel);

  // ---- scores: 16 tiles of 16x16, K=64 (2 WMMAs each)
  v8f sc[16];
#pragma unroll
  for (int j = 0; j < 16; ++j) {
#pragma unroll
    for (int i = 0; i < 8; ++i) sc[j][i] = 0.0f;
    const u32* kp = (const u32*)&KV[(j * 16 + r) * 64];
    v16bf b0 = frag16(kp, ksel);
    v16bf b1 = frag16(kp + 16, ksel);
    sc[j] = __builtin_amdgcn_wmma_f32_16x16x32_bf16(false, aq0, false, b0, (short)0, sc[j], false, false);
    sc[j] = __builtin_amdgcn_wmma_f32_16x16x32_bf16(false, aq1, false, b1, (short)0, sc[j], false, false);
  }
  __syncthreads();

  // ---- reuse LDS: stage V transposed Vt[d][m]
  {
    const bf16* vsrc = qkv + (size_t)(b * NP + tid) * 1536 + 1024 + hh * 64;
#pragma unroll
    for (int d = 0; d < 64; ++d) KV[d * 256 + tid] = vsrc[d];
  }
  __syncthreads();

  // ---- scale, diagonal mask, row max
  float rmax[8], rsum[8];
#pragma unroll
  for (int i = 0; i < 8; ++i) rmax[i] = -3.0e38f;
#pragma unroll
  for (int j = 0; j < 16; ++j)
#pragma unroll
    for (int i = 0; i < 8; ++i) {
      float v = sc[j][i] * tscale;
      int   qr = qrow0 + i + 8 * hlf;
      int   cc = j * 16 + nl;
      if (qr == cc) v = -3.0e38f;
      sc[j][i] = v;
      rmax[i]  = fmaxf(rmax[i], v);
    }
#pragma unroll
  for (int off = 8; off; off >>= 1)
#pragma unroll
    for (int i = 0; i < 8; ++i) rmax[i] = fmaxf(rmax[i], __shfl_xor(rmax[i], off, 32));

  // ---- exp + row sum (exp stored back into sc)
#pragma unroll
  for (int i = 0; i < 8; ++i) rsum[i] = 0.0f;
#pragma unroll
  for (int j = 0; j < 16; ++j)
#pragma unroll
    for (int i = 0; i < 8; ++i) {
      float e = __expf(sc[j][i] - rmax[i]);
      sc[j][i] = e;
      rsum[i] += e;
    }
#pragma unroll
  for (int off = 8; off; off >>= 1)
#pragma unroll
    for (int i = 0; i < 8; ++i) rsum[i] += __shfl_xor(rsum[i], off, 32);
  float rinv[8];
#pragma unroll
  for (int i = 0; i < 8; ++i) rinv[i] = 1.0f / rsum[i];

  // ---- O = P @ V, K=256 in 8 steps of 32; P normalized on the fly
  v8f oc[4];
#pragma unroll
  for (int dt = 0; dt < 4; ++dt)
#pragma unroll
    for (int i = 0; i < 8; ++i) oc[dt][i] = 0.0f;

  for (int js = 0; js < 8; ++js) {
#pragma unroll
    for (int t2 = 0; t2 < 2; ++t2) {
      int j = 2 * js + t2;
#pragma unroll
      for (int i = 0; i < 8; ++i) {
        int m = i + 8 * hlf;
        ps[wid][m][t2 * 16 + nl] = (bf16)(sc[j][i] * rinv[i]);
      }
    }
    asm volatile("s_wait_dscnt 0x0" ::: "memory");  // lane-crossed LDS RAW
    v16bf ap = frag16((const u32*)&ps[wid][r][0], ksel);
#pragma unroll
    for (int dt = 0; dt < 4; ++dt) {
      const u32* vp = (const u32*)&KV[(dt * 16 + r) * 256];
      v16bf bv = frag16(vp + js * 16, ksel);
      oc[dt] = __builtin_amdgcn_wmma_f32_16x16x32_bf16(false, ap, false, bv, (short)0, oc[dt], false, false);
    }
  }

  // ---- store O as bf16 (A-input of out-projection GEMM)
#pragma unroll
  for (int dt = 0; dt < 4; ++dt)
#pragma unroll
    for (int i = 0; i < 8; ++i) {
      int m   = i + 8 * hlf;
      int row = b * NP + qrow0 + m;
      int col = hh * 64 + dt * 16 + nl;
      outb[(size_t)row * DIM + col] = (bf16)oc[dt][i];
    }
}

// ---------------------------------------------------------------------------
// LayerNorm: one row (512) per block of 256 threads; writes bf16.
// ---------------------------------------------------------------------------
__global__ void __launch_bounds__(256) ln_kernel(
    const float* __restrict__ x, const float* __restrict__ w,
    const float* __restrict__ bb, bf16* __restrict__ y) {
  __shared__ float red[8];
  const int row = blockIdx.x, tid = threadIdx.x, lane = tid & 31, wid = tid >> 5;
  const float* xr = x + (size_t)row * DIM;
  float x0 = xr[tid], x1 = xr[tid + 256];

  float s = x0 + x1;
#pragma unroll
  for (int off = 16; off; off >>= 1) s += __shfl_xor(s, off, 32);
  if (lane == 0) red[wid] = s;
  __syncthreads();
  float tot = 0.0f;
#pragma unroll
  for (int i = 0; i < 8; ++i) tot += red[i];
  float mean = tot * (1.0f / DIM);
  __syncthreads();

  float d0 = x0 - mean, d1 = x1 - mean;
  float v = d0 * d0 + d1 * d1;
#pragma unroll
  for (int off = 16; off; off >>= 1) v += __shfl_xor(v, off, 32);
  if (lane == 0) red[wid] = v;
  __syncthreads();
  tot = 0.0f;
#pragma unroll
  for (int i = 0; i < 8; ++i) tot += red[i];
  float rstd = rsqrtf(tot * (1.0f / DIM) + EPSV);

  y[(size_t)row * DIM + tid]       = (bf16)(d0 * rstd * w[tid] + bb[tid]);
  y[(size_t)row * DIM + tid + 256] = (bf16)(d1 * rstd * w[tid + 256] + bb[tid + 256]);
}

// h[b,n,:] = x[b,0,:] + pos_emb[idx[b,n],:]
__global__ void __launch_bounds__(256) embed_kernel(
    const float* __restrict__ x, const int* __restrict__ idx,
    const float* __restrict__ pos, float* __restrict__ h) {
  const int row = blockIdx.x, tid = threadIdx.x;
  const int b = row >> 8;
  const int p = idx[row];
  h[(size_t)row * DIM + tid]       = x[b * DIM + tid]       + pos[(size_t)p * DIM + tid];
  h[(size_t)row * DIM + tid + 256] = x[b * DIM + tid + 256] + pos[(size_t)p * DIM + tid + 256];
}

__global__ void __launch_bounds__(256) cvt_kernel(const float* __restrict__ s,
                                                  bf16* __restrict__ d, int n) {
  int i = blockIdx.x * 256 + threadIdx.x;
  if (i < n) d[i] = (bf16)s[i];
}

// At[(b*256+s), c] = h_flat[b, c*256 + s]  (raw .view (B,512,16,16) gather)
__global__ void __launch_bounds__(256) ct1_gather(const float* __restrict__ h,
                                                  bf16* __restrict__ At) {
  int i = blockIdx.x * 256 + threadIdx.x;     // over 16384*512
  int row = i >> 9, c = i & 511;
  int b = row >> 8, s2 = row & 255;
  At[i] = (bf16)h[(size_t)b * 131072 + c * 256 + s2];
}

// pad ct2 weights (64x48, [c][o*16+p*4+q]) to 64x64 bf16 (zeros in cols 48..63)
__global__ void __launch_bounds__(256) pad_w2(const float* __restrict__ w2,
                                              bf16* __restrict__ wp) {
  int i = blockIdx.x * 256 + threadIdx.x;     // over 64*64
  int k = i >> 6, n = i & 63;
  wp[i] = (bf16)(n < 48 ? w2[k * 48 + n] : 0.0f);
}

// ---------------------------------------------------------------------------
// ct2 via WMMA: C(262144 x 48) = A2(262144 x 64) * Wp(64 x 64, cols 48+ zero).
// 8 waves per block, each wave a 16-row tile, 3 active n-tiles, K=64.
// Epilogue: bias + LeakyReLU + ConvTranspose(4,4) pixel scatter to d_out.
// ---------------------------------------------------------------------------
__global__ void __launch_bounds__(256) ct2_wmma(
    const bf16* __restrict__ A2, const bf16* __restrict__ Wp,
    const float* __restrict__ b2, float* __restrict__ out) {
  __shared__ __align__(16) bf16 Wt[64][64];   // [n][k]
  const int tid = threadIdx.x, lane = tid & 31, wid = tid >> 5;
  const int ksel = lane >> 4, r = lane & 15;

  // stage Wp (64x64, [k][n]) transposed into Wt[n][k]
#pragma unroll
  for (int it = 0; it < 2; ++it) {
    int c = tid + it * 256;      // 512 chunks of 8 cols
    int k = c & 63;
    int nch = c >> 6;            // 0..7
    union { uint4 u4; bf16 h[8]; } tmp;
    tmp.u4 = *(const uint4*)(Wp + k * 64 + nch * 8);
#pragma unroll
    for (int i = 0; i < 8; ++i) Wt[nch * 8 + i][k] = tmp.h[i];
  }
  __syncthreads();

  const int wrow0 = (blockIdx.x * 8 + wid) * 16;
  const u32* ap_ = (const u32*)(A2 + (size_t)(wrow0 + r) * 64);
  v16bf a0 = frag16(ap_, ksel);
  v16bf a1 = frag16(ap_ + 16, ksel);

  v8f acc[3];
#pragma unroll
  for (int tn = 0; tn < 3; ++tn) {
#pragma unroll
    for (int i = 0; i < 8; ++i) acc[tn][i] = 0.0f;
    const u32* wp_ = (const u32*)&Wt[tn * 16 + r][0];
    v16bf b0 = frag16(wp_, ksel);
    v16bf b1 = frag16(wp_ + 16, ksel);
    acc[tn] = __builtin_amdgcn_wmma_f32_16x16x32_bf16(false, a0, false, b0, (short)0, acc[tn], false, false);
    acc[tn] = __builtin_amdgcn_wmma_f32_16x16x32_bf16(false, a1, false, b1, (short)0, acc[tn], false, false);
  }

  const int hlf = lane >> 4, nl = lane & 15;
#pragma unroll
  for (int tn = 0; tn < 3; ++tn) {          // col = tn*16+nl; o = tn
    float bb = b2[tn];
#pragma unroll
    for (int i = 0; i < 8; ++i) {
      int row = wrow0 + i + 8 * hlf;        // b*4096 + i2*64 + j2
      int b = row >> 12, s2 = row & 4095;
      int i2 = s2 >> 6, j2 = s2 & 63;
      int y = 4 * i2 + (nl >> 2), x = 4 * j2 + (nl & 3);
      float g = acc[tn][i] + bb;
      g = (g >= 0.0f) ? g : 0.2f * g;
      out[(size_t)b * 196608 + tn * 65536 + y * 256 + x] = g;
    }
  }
}

// ---------------------------------------------------------------------------
extern "C" void kernel_launch(void* const* d_in, const int* in_sizes, int n_in,
                              void* d_out, int out_size, void* d_ws, size_t ws_size,
                              hipStream_t stream) {
  (void)in_sizes; (void)n_in; (void)out_size; (void)ws_size;
  const float* x    = (const float*)d_in[0];
  const int*   ridx = (const int*)d_in[1];
  const float* pos  = (const float*)d_in[2];
  const float* ln1w = (const float*)d_in[3];
  const float* ln1b = (const float*)d_in[4];
  const float* qkvw = (const float*)d_in[5];
  const float* temp = (const float*)d_in[6];
  const float* outw = (const float*)d_in[7];
  const float* outbp= (const float*)d_in[8];
  const float* ln2w = (const float*)d_in[9];
  const float* ln2b = (const float*)d_in[10];
  const float* ffw1 = (const float*)d_in[11];
  const float* ffb1 = (const float*)d_in[12];
  const float* ffw2 = (const float*)d_in[13];
  const float* ffb2 = (const float*)d_in[14];
  const float* ct1w = (const float*)d_in[15];
  const float* ct1b = (const float*)d_in[16];
  const float* ct2w = (const float*)d_in[17];
  const float* ct2b = (const float*)d_in[18];
  float* out = (float*)d_out;

  // ---- workspace bump allocator (256B aligned)
  char*  wsp = (char*)d_ws;
  size_t off = 0;
  auto alloc = [&](size_t bytes) -> char* {
    char* p = wsp + off;
    off += (bytes + 255) & ~(size_t)255;
    return p;
  };
  const size_t WQKV_E = (size_t)DEPTH * DIM * 3 * NHEAD * DHEAD;  // 4718592
  const size_t WOUT_E = (size_t)DEPTH * DIM * DIM;                // 1572864
  const size_t WFF1_E = (size_t)DEPTH * DIM * MLPD;               // 6291456
  const size_t WFF2_E = (size_t)DEPTH * MLPD * DIM;               // 6291456
  const size_t WCT1_E = (size_t)DIM * 1024;                       // 524288
  const size_t H_E    = (size_t)MROWS * DIM;                      // 8388608
  const size_t QKV_E  = (size_t)MROWS * 1536;
  const size_t FF1_E  = (size_t)MROWS * MLPD;
  const size_t A2_E   = (size_t)BATCH * 4096 * 64;

  bf16*  wqkv  = (bf16*)alloc(WQKV_E * 2);
  bf16*  wout  = (bf16*)alloc(WOUT_E * 2);
  bf16*  wff1  = (bf16*)alloc(WFF1_E * 2);
  bf16*  wff2  = (bf16*)alloc(WFF2_E * 2);
  bf16*  wct1  = (bf16*)alloc(WCT1_E * 2);
  bf16*  wp2   = (bf16*)alloc(64 * 64 * 2);
  float* hbuf  = (float*)alloc(H_E * 4);
  bf16*  lnbuf = (bf16*)alloc(H_E * 2);
  bf16*  qkvb  = (bf16*)alloc(QKV_E * 2);
  bf16*  attb  = (bf16*)alloc(H_E * 2);
  bf16*  ff1b  = (bf16*)alloc(FF1_E * 2);
  bf16*  Atb   = (bf16*)alloc(H_E * 2);
  bf16*  A2b   = (bf16*)alloc(A2_E * 2);

  const int thr = 256;
  // ---- weights fp32 -> bf16 (+ ct2 pad)
  cvt_kernel<<<(int)((WQKV_E + 255) / 256), thr, 0, stream>>>(qkvw, wqkv, (int)WQKV_E);
  cvt_kernel<<<(int)((WOUT_E + 255) / 256), thr, 0, stream>>>(outw, wout, (int)WOUT_E);
  cvt_kernel<<<(int)((WFF1_E + 255) / 256), thr, 0, stream>>>(ffw1, wff1, (int)WFF1_E);
  cvt_kernel<<<(int)((WFF2_E + 255) / 256), thr, 0, stream>>>(ffw2, wff2, (int)WFF2_E);
  cvt_kernel<<<(int)((WCT1_E + 255) / 256), thr, 0, stream>>>(ct1w, wct1, (int)WCT1_E);
  pad_w2<<<16, thr, 0, stream>>>(ct2w, wp2);

  // ---- token embedding
  embed_kernel<<<MROWS, thr, 0, stream>>>(x, ridx, pos, hbuf);

  // ---- transformer layers
  for (int L = 0; L < DEPTH; ++L) {
    ln_kernel<<<MROWS, thr, 0, stream>>>(hbuf, ln1w + L * DIM, ln1b + L * DIM, lnbuf);
    gemm_bf16<0><<<dim3(12, 128), thr, 0, stream>>>(
        lnbuf, wqkv + (size_t)L * DIM * 1536, MROWS, 1536, DIM, nullptr, qkvb, nullptr);
    attn_kernel<<<BATCH * NHEAD * 2, thr, 0, stream>>>(qkvb, temp, L, attb);
    gemm_bf16<1><<<dim3(4, 128), thr, 0, stream>>>(
        attb, wout + (size_t)L * DIM * DIM, MROWS, DIM, DIM, hbuf, nullptr, outbp + L * DIM);
    ln_kernel<<<MROWS, thr, 0, stream>>>(hbuf, ln2w + L * DIM, ln2b + L * DIM, lnbuf);
    gemm_bf16<2><<<dim3(16, 128), thr, 0, stream>>>(
        lnbuf, wff1 + (size_t)L * DIM * MLPD, MROWS, MLPD, DIM, nullptr, ff1b, ffb1 + L * MLPD);
    gemm_bf16<1><<<dim3(4, 128), thr, 0, stream>>>(
        ff1b, wff2 + (size_t)L * MLPD * DIM, MROWS, DIM, MLPD, hbuf, nullptr, ffb2 + L * DIM);
  }

  // ---- decoder: ct1 as GEMM (16384x1024x512) with scatter epilogue,
  //      then ct2 as WMMA GEMM (262144x48x64) with pixel-scatter epilogue
  ct1_gather<<<(int)(H_E / 256), thr, 0, stream>>>(hbuf, Atb);
  gemm_bf16<3><<<dim3(8, 128), thr, 0, stream>>>(
      Atb, wct1, MROWS, 1024, DIM, nullptr, A2b, ct1b);
  ct2_wmma<<<2048, thr, 0, stream>>>(A2b, wp2, ct2b, out);
}
